// FusedAttention_2534030704871
// MI455X (gfx1250) — compile-verified, matrix operand order checked by reading, not run on hardware
//
#include <hip/hip_runtime.h>
#include <hip/hip_bf16.h>

typedef __attribute__((ext_vector_type(16))) _Float16 v16h;
typedef __attribute__((ext_vector_type(8)))  _Float16 v8h;
typedef __attribute__((ext_vector_type(8)))  float    v8f;

#define BATCH 4
#define CCH   256
#define NPIX  4096
#define CQKD  32
#define CGRP  128   // channels per wave (8 WMMA fragments)

// Lane <-> lane^16 exchange with a single ds_swizzle_b32 (SWAPX16 pattern:
// xor_mask=0x10, and_mask=0x1f -> offset 0x401F).
__device__ __forceinline__ float swap16(float x) {
  return __builtin_bit_cast(float,
      __builtin_amdgcn_ds_swizzle(__builtin_bit_cast(int, x), 0x401F));
}

// ---------------------------------------------------------------------------
// Projection kernel: out[o,i] = sum_c W[o,c] * x[b,c,i] + bias[o], f16 output.
// tmode=1 -> out layout (B, N, CQKD)  (transposed, for f/g WMMA fragments)
// tmode=0 -> out layout (B, Ochan, N) (direct, for hv WMMA fragments)
// ---------------------------------------------------------------------------
__global__ __launch_bounds__(256) void proj_kernel(
    const float* __restrict__ x, const float* __restrict__ W,
    const float* __restrict__ bias, _Float16* __restrict__ out,
    int Ochan, int tmode)
{
  __shared__ float xs[CCH][64];
  const int b   = blockIdx.y;
  const int i0  = blockIdx.x * 64;
  const int tid = threadIdx.x;
  const float* xb = x + (size_t)b * CCH * NPIX;

  for (int e = tid; e < CCH * 64; e += 256) {
    int c = e >> 6, ii = e & 63;
    xs[c][ii] = xb[(size_t)c * NPIX + i0 + ii];
  }
  __syncthreads();

  const int ii = tid & 63;
  const int ob = tid >> 6;                       // 0..3
  for (int o = ob; o < Ochan; o += 4) {          // o uniform within a wave
    float acc = bias[o];
    const float* wrow = W + (size_t)o * CCH;
    #pragma unroll 8
    for (int c = 0; c < CCH; ++c)
      acc = fmaf(wrow[c], xs[c][ii], acc);
    const int i = i0 + ii;
    if (tmode)
      out[((size_t)b * NPIX + i) * CQKD + o] = (_Float16)acc;
    else
      out[((size_t)b * Ochan + o) * NPIX + i] = (_Float16)acc;
  }
}

// ---------------------------------------------------------------------------
// Flash-attention kernel, S^T formulation.
// One wave = 16 queries x 128 output channels (8 accumulator fragments).
// Block = 4 waves = 2 query tiles x 2 channel halves (32 queries, 256 ch).
// Per 32-key step: 2 score WMMAs (S^T tiles), in-lane online softmax with a
// single lane^16 exchange, then the exp'd fragment is ALREADY in B-fragment
// layout for the 8 PV WMMAs (O^T = hv x P^T). No LDS, no transposes.
// ---------------------------------------------------------------------------
__global__ __launch_bounds__(128) void flashattn_kernel(
    const _Float16* __restrict__ fT,   // (B, N, 32)
    const _Float16* __restrict__ gT,   // (B, N, 32)
    const _Float16* __restrict__ hv,   // (B, C, N)
    const float*    __restrict__ vin,  // (B, C, N)
    const float*    __restrict__ gamma,
    float*          __restrict__ outp) // (B, C, N)
{
  const int lane = threadIdx.x & 31;
  const int wv   = threadIdx.x >> 5;             // 0..3
  const int b    = blockIdx.y;
  const int q0   = blockIdx.x * 32 + (wv >> 1) * 16;  // query tile
  const int c0   = (wv & 1) * CGRP;                   // channel half
  const int half = lane >> 4;            // 0: lanes 0-15, 1: lanes 16-31
  const int lid  = lane & 15;
  const int kb1  = half ? 8 : 0;         // K-base for A/B VGPRs 0..3
  const int kb2  = kb1 + 16;             // K-base for A/B VGPRs 4..7

  // B fragment: f (K=32 channels x N=16 queries). Lane col = query q0+lid.
  v16h bF;
  {
    const _Float16* fp = fT + ((size_t)b * NPIX + (q0 + lid)) * CQKD;
    v8h lo = *(const v8h*)(fp + kb1);
    v8h hi = *(const v8h*)(fp + kb2);
    #pragma unroll
    for (int t = 0; t < 8; ++t) { bF[t] = lo[t]; bF[8 + t] = hi[t]; }
  }

  v8f Oacc[8] = {};                      // O^T: rows=channels, cols=queries
  float m = -1e30f, l = 0.0f;            // per-query stats (one query / lane)

  const _Float16* gb = gT + (size_t)b * NPIX * CQKD;
  const _Float16* hb = hv + (size_t)b * CCH * NPIX;

  for (int j0 = 0; j0 < NPIX; j0 += 32) {
    // --- A fragments of g: key rows [j0,j0+16) and [j0+16,j0+32) ---
    v16h aG0, aG1;
    {
      const _Float16* gp0 = gb + (size_t)(j0 + lid) * CQKD;
      const _Float16* gp1 = gb + (size_t)(j0 + 16 + lid) * CQKD;
      v8h a0 = *(const v8h*)(gp0 + kb1);
      v8h a1 = *(const v8h*)(gp0 + kb2);
      v8h b0 = *(const v8h*)(gp1 + kb1);
      v8h b1 = *(const v8h*)(gp1 + kb2);
      #pragma unroll
      for (int t = 0; t < 8; ++t) {
        aG0[t] = a0[t]; aG0[8 + t] = a1[t];
        aG1[t] = b0[t]; aG1[8 + t] = b1[t];
      }
    }
    __builtin_prefetch(gb + (size_t)(j0 + 32 + lid) * CQKD, 0, 1);

    // S^T tiles: D[key, query]; lane holds query col lid, keys r+8*half(+16)
    v8f zero = {};
    v8f s0 = __builtin_amdgcn_wmma_f32_16x16x32_f16(false, aG0, false, bF,
                                                    (short)0, zero, false, false);
    v8f s1 = __builtin_amdgcn_wmma_f32_16x16x32_f16(false, aG1, false, bF,
                                                    (short)0, zero, false, false);

    // --- online softmax: in-lane over 16 keys + one lane^16 exchange ---
    float t = fmaxf(s0[0], s1[0]);
    #pragma unroll
    for (int r = 1; r < 8; ++r) t = fmaxf(t, fmaxf(s0[r], s1[r]));
    t = fmaxf(t, swap16(t));
    const float mn = fmaxf(m, t);
    const float alpha = __expf(m - mn);
    m = mn;

    float p0[8], p1[8], rs = 0.0f;
    #pragma unroll
    for (int r = 0; r < 8; ++r) {
      p0[r] = __expf(s0[r] - mn);
      p1[r] = __expf(s1[r] - mn);
      rs += p0[r] + p1[r];
    }
    rs += swap16(rs);
    l = l * alpha + rs;

    // P^T fragment is directly in B layout for the PV WMMA: lane (half,lid)
    // holds query col lid with keys {8h+t} (VGPRs 0-3) and {16+8h+t} (4-7).
    v16h bP;
    #pragma unroll
    for (int t2 = 0; t2 < 8; ++t2) {
      bP[t2]     = (_Float16)p0[t2];
      bP[8 + t2] = (_Float16)p1[t2];
    }

    #pragma unroll
    for (int g = 0; g < 8; ++g) {
      #pragma unroll
      for (int r = 0; r < 8; ++r) Oacc[g][r] *= alpha;
    }

    // --- 8 PV WMMAs: O^T[c, q] += sum_k hv[c, j0+k] * P^T[k, q] ---
    #pragma unroll
    for (int g = 0; g < 8; ++g) {
      const _Float16* hp = hb + (size_t)(c0 + g * 16 + lid) * NPIX + j0;
      v8h lo = *(const v8h*)(hp + kb1);
      v8h hi = *(const v8h*)(hp + kb2);
      v16h aH;
      #pragma unroll
      for (int t2 = 0; t2 < 8; ++t2) { aH[t2] = lo[t2]; aH[8 + t2] = hi[t2]; }
      Oacc[g] = __builtin_amdgcn_wmma_f32_16x16x32_f16(false, aH, false, bP,
                                                       (short)0, Oacc[g],
                                                       false, false);
    }
  }

  // --- epilogue: out[b,c,q0+lid] = gamma * O^T/l + v;  contiguous stores ---
  const float gm   = gamma[0];
  const float rinv = 1.0f / l;
  const float* vb  = vin  + (size_t)b * CCH * NPIX;
  float*       ob  = outp + (size_t)b * CCH * NPIX;
  #pragma unroll
  for (int g = 0; g < 8; ++g) {
    #pragma unroll
    for (int r = 0; r < 8; ++r) {
      const int ch = c0 + g * 16 + r + (half << 3);
      const size_t idx = (size_t)ch * NPIX + q0 + lid;
      ob[idx] = gm * Oacc[g][r] * rinv + vb[idx];
    }
  }
}

// ---------------------------------------------------------------------------
extern "C" void kernel_launch(void* const* d_in, const int* in_sizes, int n_in,
                              void* d_out, int out_size, void* d_ws, size_t ws_size,
                              hipStream_t stream) {
  const float* q     = (const float*)d_in[0];
  const float* k     = (const float*)d_in[1];
  const float* v     = (const float*)d_in[2];
  const float* Wq    = (const float*)d_in[3];
  const float* bq    = (const float*)d_in[4];
  const float* Wk    = (const float*)d_in[5];
  const float* bk    = (const float*)d_in[6];
  const float* Wv    = (const float*)d_in[7];
  const float* bv    = (const float*)d_in[8];
  const float* gamma = (const float*)d_in[9];
  float* out = (float*)d_out;

  _Float16* fT = (_Float16*)d_ws;                       // (B, N, 32)  1 MB
  _Float16* gT = fT + (size_t)BATCH * NPIX * CQKD;      // (B, N, 32)  1 MB
  _Float16* hv = gT + (size_t)BATCH * NPIX * CQKD;      // (B, C, N)   8 MB

  dim3 pgrid(NPIX / 64, BATCH);
  proj_kernel<<<pgrid, 256, 0, stream>>>(q, Wq, bq, fT, CQKD, 1);
  proj_kernel<<<pgrid, 256, 0, stream>>>(k, Wk, bk, gT, CQKD, 1);
  proj_kernel<<<pgrid, 256, 0, stream>>>(v, Wv, bv, hv, CCH, 0);

  dim3 agrid(NPIX / 32, BATCH);
  flashattn_kernel<<<agrid, 128, 0, stream>>>(fT, gT, hv, v, gamma, out);
}
